// Seq2SeqLSTM_42674795053327
// MI455X (gfx1250) — matrix operand order repaired and checked by reading, not compiled, not run on hardware
//
#include <hip/hip_runtime.h>
#include <hip/hip_bf16.h>
#include <cstdint>
#include <cstddef>

// ---------------------------------------------------------------------------
// Seq2Seq LSTM for MI455X (gfx1250, wave32, WMMA).
// All GEMMs (x_gates precompute, recurrent h@Whh^T, vocab FC) go through
// v_wmma_f32_16x16x32_f16 with f32 accumulation. Per-wave tile is 32x64
// (8 accumulators): halves B-matrix (weight) re-reads vs 16x64 and gives
// 8 WMMAs per ~12 b128 loads. fc_W (64MB f16) is L2-resident (192MB L2).
// ---------------------------------------------------------------------------

typedef __attribute__((ext_vector_type(16))) _Float16 v16h;
typedef __attribute__((ext_vector_type(8)))  _Float16 v8h;
typedef __attribute__((ext_vector_type(8)))  float    v8f;

#define B_  32
#define T_  128
#define V_  32000
#define E_  512
#define H_  1024
#define G_  4096   // 4*H

// ---------------- utility kernels ----------------

__global__ __launch_bounds__(256) void k_f32_to_f16(const float* __restrict__ s,
                                                    _Float16* __restrict__ d, size_t n)
{
    size_t i = (size_t)blockIdx.x * blockDim.x + threadIdx.x;
    size_t stride = (size_t)gridDim.x * blockDim.x;
    for (; i < n; i += stride) d[i] = (_Float16)s[i];
}

__global__ __launch_bounds__(256) void k_zero_u32(uint32_t* __restrict__ p, size_t n)
{
    size_t i = (size_t)blockIdx.x * blockDim.x + threadIdx.x;
    size_t stride = (size_t)gridDim.x * blockDim.x;
    for (; i < n; i += stride) p[i] = 0u;
}

// Gather embedding rows, convert f32 -> f16, time-major [T,B,E].
// shift=1: decoder teacher forcing (token 0 at t=0, else targets[b][t-1]).
__global__ __launch_bounds__(128) void k_embed_gather(const int* __restrict__ tok,
                                                      const float* __restrict__ emb,
                                                      _Float16* __restrict__ dst, int shift)
{
    int t = blockIdx.x, b = blockIdx.y;
    int tk = shift ? ((t == 0) ? 0 : tok[b * T_ + (t - 1)]) : tok[b * T_ + t];
    const float* src = emb + (size_t)tk * E_;
    _Float16* d = dst + ((size_t)t * B_ + b) * E_;
    for (int e = threadIdx.x; e < E_; e += blockDim.x) d[e] = (_Float16)src[e];
}

// ---------------- WMMA GEMM ----------------
// C[M x N] = A[M x K] * W[N x K]^T (+ Cin[M x N]) (+ bias[N]),  f16 in, f32 out.
// One wave computes a 32x64 output tile: 2 A fragments x 4 B fragments ->
// 8 wmma accumulators per k-step. Block = 128 threads = 4 waves covering
// 32 rows x 256 cols. M % 32 == 0, N % 256 == 0, K % 32 == 0 (all shapes here).
// swapBT != 0 remaps output row r = t*B+b -> b*T+t (emit [B,T,V] logits).
__global__ __launch_bounds__(128) void k_wmma_gemm(const _Float16* __restrict__ A,
                                                   const _Float16* __restrict__ W,
                                                   const float* __restrict__ bias,
                                                   const float* __restrict__ Cin,
                                                   float* __restrict__ C,
                                                   int M, int N, int K, int swapBT)
{
    const int lane = threadIdx.x & 31;
    const int wave = threadIdx.x >> 5;
    const int m0   = blockIdx.y << 5;                // 32 rows per tile
    const int n0   = (blockIdx.x * 4 + wave) << 6;   // 64 cols per wave
    const int nc   = lane & 15;

    v8f acc[8] = {};   // acc[mi*4+j]: mi = row-half (0/1), j = 16-col group

    // A fragment (16x32 f16): lanes 0-15 row M=lane hold K k..k+15;
    // lanes 16-31 same rows hold K k+16..k+31.
    const int aOff = (lane < 16) ? 0 : 16;
    // B fragment (32x16 f16, column-major == rows of W): lanes 0-15 col N=n hold
    // K k..k+15; lanes 16-31 hold K k+16..k+31.
    const int bOff = (lane < 16) ? 0 : 16;

    const _Float16* a0 = A + (size_t)(m0 +  0 + nc) * K + aOff;
    const _Float16* a1 = A + (size_t)(m0 + 16 + nc) * K + aOff;
    const _Float16* wp0 = W + (size_t)(n0 +  0 + nc) * K + bOff;
    const _Float16* wp1 = W + (size_t)(n0 + 16 + nc) * K + bOff;
    const _Float16* wp2 = W + (size_t)(n0 + 32 + nc) * K + bOff;
    const _Float16* wp3 = W + (size_t)(n0 + 48 + nc) * K + bOff;

    for (int k = 0; k < K; k += 32) {
        // prefetch weight streams four k-steps (128 elements) ahead into L2/WGP$ (global_prefetch_b8)
        __builtin_prefetch((const void*)(wp0 + k + 128), 0, 1);
        __builtin_prefetch((const void*)(wp1 + k + 128), 0, 1);
        __builtin_prefetch((const void*)(wp2 + k + 128), 0, 1);
        __builtin_prefetch((const void*)(wp3 + k + 128), 0, 1);

        v8h lo0 = *(const v8h*)(a0 + k);
        v8h hi0 = *(const v8h*)(a0 + k + 8);
        v8h lo1 = *(const v8h*)(a1 + k);
        v8h hi1 = *(const v8h*)(a1 + k + 8);
        v16h afA, afB;
#pragma unroll
        for (int i = 0; i < 8; ++i) {
            afA[i] = lo0[i]; afA[i + 8] = hi0[i];
            afB[i] = lo1[i]; afB[i + 8] = hi1[i];
        }
        v16h b0 = *(const v16h*)(wp0 + k);
        v16h b1 = *(const v16h*)(wp1 + k);
        v16h b2 = *(const v16h*)(wp2 + k);
        v16h b3 = *(const v16h*)(wp3 + k);
        acc[0] = __builtin_amdgcn_wmma_f32_16x16x32_f16(false, afA, false, b0, (short)0, acc[0], false, false);
        acc[4] = __builtin_amdgcn_wmma_f32_16x16x32_f16(false, afB, false, b0, (short)0, acc[4], false, false);
        acc[1] = __builtin_amdgcn_wmma_f32_16x16x32_f16(false, afA, false, b1, (short)0, acc[1], false, false);
        acc[5] = __builtin_amdgcn_wmma_f32_16x16x32_f16(false, afB, false, b1, (short)0, acc[5], false, false);
        acc[2] = __builtin_amdgcn_wmma_f32_16x16x32_f16(false, afA, false, b2, (short)0, acc[2], false, false);
        acc[6] = __builtin_amdgcn_wmma_f32_16x16x32_f16(false, afB, false, b2, (short)0, acc[6], false, false);
        acc[3] = __builtin_amdgcn_wmma_f32_16x16x32_f16(false, afA, false, b3, (short)0, acc[3], false, false);
        acc[7] = __builtin_amdgcn_wmma_f32_16x16x32_f16(false, afB, false, b3, (short)0, acc[7], false, false);
    }

    // C/D layout: VGPR r, lanes 0-15 -> M = base + r, lanes 16-31 -> M = base+8+r;
    // N = n-tile base + (lane & 15).
    const int rhalf = (lane < 16) ? 0 : 8;
#pragma unroll
    for (int mi = 0; mi < 2; ++mi) {
        const int rbase = m0 + mi * 16 + rhalf;
#pragma unroll
        for (int j = 0; j < 4; ++j) {
            int n = n0 + j * 16 + nc;
            float bv = bias ? bias[n] : 0.0f;
#pragma unroll
            for (int r = 0; r < 8; ++r) {
                int row = rbase + r;
                float v = acc[mi * 4 + j][r] + bv;
                if (Cin) v += Cin[(size_t)row * N + n];
                size_t orow = swapBT ? ((size_t)(row % B_) * T_ + (size_t)(row / B_))
                                     : (size_t)row;
                C[orow * (size_t)N + n] = v;
            }
        }
    }
}

// ---------------- LSTM cell (elementwise, f32) ----------------
// gates [B, 4H] blocks (i,f,g,o); updates c, writes h as f16 (next-step GEMM
// input) and into the layer-output history hs[t].
__device__ __forceinline__ float sigmoidf_(float x) { return 1.0f / (1.0f + __expf(-x)); }

__global__ __launch_bounds__(256) void k_lstm_cell(const float* __restrict__ gates,
                                                   float* __restrict__ c,
                                                   _Float16* __restrict__ hf16,
                                                   _Float16* __restrict__ hs_t)
{
    int idx = blockIdx.x * blockDim.x + threadIdx.x;   // B_*H_ threads
    if (idx >= B_ * H_) return;
    int b = idx >> 10;          // /H_
    int h = idx & (H_ - 1);
    const float* g = gates + (size_t)b * G_;
    float i_ = sigmoidf_(g[h]);
    float f_ = sigmoidf_(g[H_ + h]);
    float gg = tanhf(g[2 * H_ + h]);
    float o_ = sigmoidf_(g[3 * H_ + h]);
    float cn = f_ * c[idx] + i_ * gg;
    c[idx] = cn;
    float hn = o_ * tanhf(cn);
    _Float16 hh = (_Float16)hn;
    hf16[idx] = hh;
    hs_t[idx] = hh;
}

// ---------------- log-softmax over V per row, in place ----------------
__global__ __launch_bounds__(256) void k_log_softmax(float* __restrict__ out)
{
    float* p = out + (size_t)blockIdx.x * V_;
    __shared__ float red[256];
    float mx = -3.402823466e38f;
    for (int v = threadIdx.x; v < V_; v += 256) mx = fmaxf(mx, p[v]);
    red[threadIdx.x] = mx; __syncthreads();
    for (int s = 128; s > 0; s >>= 1) {
        if (threadIdx.x < s) red[threadIdx.x] = fmaxf(red[threadIdx.x], red[threadIdx.x + s]);
        __syncthreads();
    }
    mx = red[0]; __syncthreads();
    float sum = 0.0f;
    for (int v = threadIdx.x; v < V_; v += 256) sum += __expf(p[v] - mx);
    red[threadIdx.x] = sum; __syncthreads();
    for (int s = 128; s > 0; s >>= 1) {
        if (threadIdx.x < s) red[threadIdx.x] += red[threadIdx.x + s];
        __syncthreads();
    }
    float lse = mx + __logf(red[0]);
    __syncthreads();
    for (int v = threadIdx.x; v < V_; v += 256) p[v] -= lse;
}

// ---------------- host orchestration ----------------

extern "C" void kernel_launch(void* const* d_in, const int* in_sizes, int n_in,
                              void* d_out, int out_size, void* d_ws, size_t ws_size,
                              hipStream_t stream)
{
    (void)in_sizes; (void)n_in; (void)out_size; (void)ws_size;

    const int*   x       = (const int*)  d_in[0];
    const int*   targets = (const int*)  d_in[1];
    const float* emb     = (const float*)d_in[2];
    const float* encWih0 = (const float*)d_in[3];
    const float* encWhh0 = (const float*)d_in[4];
    const float* enc_b0  = (const float*)d_in[5];
    const float* encWih1 = (const float*)d_in[6];
    const float* encWhh1 = (const float*)d_in[7];
    const float* enc_b1  = (const float*)d_in[8];
    const float* decWih0 = (const float*)d_in[9];
    const float* decWhh0 = (const float*)d_in[10];
    const float* dec_b0  = (const float*)d_in[11];
    const float* decWih1 = (const float*)d_in[12];
    const float* decWhh1 = (const float*)d_in[13];
    const float* dec_b1  = (const float*)d_in[14];
    const float* fc_W    = (const float*)d_in[15];
    const float* fc_b    = (const float*)d_in[16];
    float* out = (float*)d_out;

    // ---- workspace carve (256B aligned) ----
    char* ws = (char*)d_ws;
    size_t off = 0;
    auto alloc = [&](size_t bytes) -> char* {
        char* p = ws + off;
        off += (bytes + 255) & ~(size_t)255;
        return p;
    };
    _Float16* encWih0_h = (_Float16*)alloc((size_t)G_ * E_ * 2);
    _Float16* encWhh0_h = (_Float16*)alloc((size_t)G_ * H_ * 2);
    _Float16* encWih1_h = (_Float16*)alloc((size_t)G_ * H_ * 2);
    _Float16* encWhh1_h = (_Float16*)alloc((size_t)G_ * H_ * 2);
    _Float16* decWih0_h = (_Float16*)alloc((size_t)G_ * E_ * 2);
    _Float16* decWhh0_h = (_Float16*)alloc((size_t)G_ * H_ * 2);
    _Float16* decWih1_h = (_Float16*)alloc((size_t)G_ * H_ * 2);
    _Float16* decWhh1_h = (_Float16*)alloc((size_t)G_ * H_ * 2);
    _Float16* fcW_h     = (_Float16*)alloc((size_t)V_ * H_ * 2);
    _Float16* enc_in_h  = (_Float16*)alloc((size_t)T_ * B_ * E_ * 2);
    _Float16* dec_in_h  = (_Float16*)alloc((size_t)T_ * B_ * E_ * 2);
    _Float16* hsA       = (_Float16*)alloc((size_t)T_ * B_ * H_ * 2);
    _Float16* hsB       = (_Float16*)alloc((size_t)T_ * B_ * H_ * 2);
    float*    xg        = (float*)   alloc((size_t)T_ * B_ * G_ * 4);
    float*    gatesBuf  = (float*)   alloc((size_t)B_ * G_ * 4);
    // zero-init region: stateC0, stateC1 (f32), hf16_0, hf16_1 (f16), contiguous
    char*     zreg      = alloc((size_t)B_ * H_ * (4 + 4 + 2 + 2));
    float*    stateC0 = (float*)zreg;
    float*    stateC1 = (float*)(zreg + (size_t)B_ * H_ * 4);
    _Float16* hf16_0  = (_Float16*)(zreg + (size_t)B_ * H_ * 8);
    _Float16* hf16_1  = (_Float16*)(zreg + (size_t)B_ * H_ * 10);

    // ---- weight conversion f32 -> f16 ----
    auto cvt = [&](const float* s, _Float16* d, size_t n) {
        k_f32_to_f16<<<2048, 256, 0, stream>>>(s, d, n);
    };
    cvt(encWih0, encWih0_h, (size_t)G_ * E_);
    cvt(encWhh0, encWhh0_h, (size_t)G_ * H_);
    cvt(encWih1, encWih1_h, (size_t)G_ * H_);
    cvt(encWhh1, encWhh1_h, (size_t)G_ * H_);
    cvt(decWih0, decWih0_h, (size_t)G_ * E_);
    cvt(decWhh0, decWhh0_h, (size_t)G_ * H_);
    cvt(decWih1, decWih1_h, (size_t)G_ * H_);
    cvt(decWhh1, decWhh1_h, (size_t)G_ * H_);
    cvt(fc_W,    fcW_h,     (size_t)V_ * H_);

    // zero h0/c0 states (workspace is poisoned, must init)
    k_zero_u32<<<512, 256, 0, stream>>>((uint32_t*)zreg, (size_t)B_ * H_ * 12 / 4);

    // ---- embedding gathers ----
    k_embed_gather<<<dim3(T_, B_), 128, 0, stream>>>(x,       emb, enc_in_h, 0);
    k_embed_gather<<<dim3(T_, B_), 128, 0, stream>>>(targets, emb, dec_in_h, 1);

    const dim3 blk(128);
    // One LSTM layer: precompute x_gates for whole sequence, then scan time.
    auto run_layer = [&](const _Float16* Xin, int K, const _Float16* Wih_h,
                         const float* bias, const _Float16* Whh_h,
                         float* stateC, _Float16* hf16, _Float16* hs_out) {
        // x_gates[T*B, 4H] = Xin @ Wih^T + b
        k_wmma_gemm<<<dim3(G_ / 256, (T_ * B_) / 32), blk, 0, stream>>>(
            Xin, Wih_h, bias, nullptr, xg, T_ * B_, G_, K, 0);
        for (int t = 0; t < T_; ++t) {
            // gates = x_gates[t] + h_{t-1} @ Whh^T
            k_wmma_gemm<<<dim3(G_ / 256, B_ / 32), blk, 0, stream>>>(
                hf16, Whh_h, nullptr, xg + (size_t)t * B_ * G_, gatesBuf, B_, G_, H_, 0);
            k_lstm_cell<<<(B_ * H_ + 255) / 256, 256, 0, stream>>>(
                gatesBuf, stateC, hf16, hs_out + (size_t)t * B_ * H_);
        }
    };

    // Encoder (states start at zero)
    run_layer(enc_in_h, E_, encWih0_h, enc_b0, encWhh0_h, stateC0, hf16_0, hsA);
    run_layer(hsA,      H_, encWih1_h, enc_b1, encWhh1_h, stateC1, hf16_1, hsB);
    // Decoder (states/h carried over from encoder finals: stateC*, hf16_*)
    run_layer(dec_in_h, E_, decWih0_h, dec_b0, decWhh0_h, stateC0, hf16_0, hsA);
    run_layer(hsA,      H_, decWih1_h, dec_b1, decWhh1_h, stateC1, hf16_1, hsB);

    // FC: logits[b,t,v] = dec_hs[t,b,:] . fc_W[v,:] + fc_b[v]  (row remap t*B+b -> b*T+t)
    k_wmma_gemm<<<dim3(V_ / 256, (T_ * B_) / 32), blk, 0, stream>>>(
        hsB, fcW_h, fc_b, nullptr, out, T_ * B_, V_, H_, 1);

    // log_softmax over vocab, in place on d_out
    k_log_softmax<<<B_ * T_, 256, 0, stream>>>(out);
}
